// RootedHeadProbEncoder_53824530153943
// MI455X (gfx1250) — compile-verified, matrix operand order checked by reading, not run on hardware
//
#include <hip/hip_runtime.h>

// ---------------------------------------------------------------------------
// RootedHeadProbEncoder for MI455X (gfx1250, wave32, WMMA f32<=f16 16x16x32)
//
// GEMM refactoring (triangular dist-masks folded into head-softmax outputs):
//   G_k   = nz @ T_k             P_k = G_k @ nz^T
//   Y_k   = nh_mask @ nz         Z_k = nh_mask^T @ nz
//   m_dep = sum_c Y0@T0^T + Y1@T1^T      m_head = sum_c Z0@T0 + Z1@T1
//
// Each block = 8 waves sharing one 16-row A strip. The strip is staged into
// LDS with gfx1250 async-to-LDS loads (fallback: plain b128 copy) so all 8
// waves read A via DS while streaming their private B panels from global.
// ---------------------------------------------------------------------------

typedef __attribute__((ext_vector_type(16))) _Float16 v16h;
typedef __attribute__((ext_vector_type(8)))  float    v8f;
typedef __attribute__((ext_vector_type(2)))  _Float16 h2v;
typedef __attribute__((ext_vector_type(4)))  int      v4i_t;
typedef __attribute__((address_space(1))) v4i_t* as1_v4i;
typedef __attribute__((address_space(3))) v4i_t* as3_v4i;

#define BATCH 8
#define SEQ   256
#define DM    128
#define NH    8
#define NKD   2
#define NEGV  (-1000000000.0f)

#if defined(__has_builtin)
#if __has_builtin(__builtin_amdgcn_global_load_async_to_lds_b128)
#define HAVE_ASYNC 1
#endif
#endif
#ifndef HAVE_ASYNC
#define HAVE_ASYNC 0
#endif

// Copy 16 bytes global -> LDS (per lane).
static __device__ __forceinline__ void cp16(const _Float16* g, _Float16* s) {
#if HAVE_ASYNC
    __builtin_amdgcn_global_load_async_to_lds_b128(
        (as1_v4i)(void*)(g), (as3_v4i)(void*)(s), 0, 0);
#else
    *(float4*)s = *(const float4*)g;
#endif
}

static __device__ __forceinline__ void stage_join() {
#if HAVE_ASYNC
#if __has_builtin(__builtin_amdgcn_s_wait_asynccnt)
    __builtin_amdgcn_s_wait_asynccnt(0);
#else
    asm volatile("s_wait_asynccnt 0" ::: "memory");
#endif
#endif
    __syncthreads();
}

// ---------------- WMMA fragment helpers (wave32) ----------------------------
// A fragment: 16x32 (MxK) f16 row-major, leading dim `ld` halves. B fragments
// load identically from (NxK) row-major buffers (i.e. B^T). Per the CDNA5 ISA
// 16-bit A layout: lane = 16*h + m; half t=2v+p -> K = 2*(v&3)+p+8*h+16*(v>>2).
static __device__ __forceinline__ v16h load_frag(const _Float16* base, int ld) {
    const int lane = threadIdx.x & 31;
    const int r = lane & 15;
    const int h = lane >> 4;
    const _Float16* rowp = base + r * ld;
    v16h f;
#pragma unroll
    for (int v = 0; v < 8; ++v) {
        const int k0 = ((v & 3) << 1) + (h << 3) + ((v >> 2) << 4);
        h2v p = *(const h2v*)(rowp + k0);
        f[2 * v]     = p[0];
        f[2 * v + 1] = p[1];
    }
    return f;
}

static __device__ __forceinline__ v8f wmma_f16(const v16h& a, const v16h& b, const v8f& c) {
    return __builtin_amdgcn_wmma_f32_16x16x32_f16(false, a, false, b, (short)0, c,
                                                  false, false);
}

// D tile layout: element (m,n) with m = 8*(lane>>4) + r, n = lane&15.
static __device__ __forceinline__ void store_tile_f32(float* base, int ld, const v8f& acc) {
    const int lane = threadIdx.x & 31;
    const int nn = lane & 15;
    const int h = lane >> 4;
#pragma unroll
    for (int r = 0; r < 8; ++r) base[(8 * h + r) * ld + nn] = acc[r];
}

static __device__ __forceinline__ void store_tile_f16(_Float16* base, int ld, const v8f& acc) {
    const int lane = threadIdx.x & 31;
    const int nn = lane & 15;
    const int h = lane >> 4;
#pragma unroll
    for (int r = 0; r < 8; ++r) base[(8 * h + r) * ld + nn] = (_Float16)acc[r];
}

// ---------------- small utility kernels -------------------------------------

__global__ void k_zero(float* p, int n) {
    int i = blockIdx.x * blockDim.x + threadIdx.x;
    if (i < n) p[i] = 0.0f;
}

// ternary (k,a,b,c) f32 -> Tab[k][c][a][b] and Tba[k][c][b][a] in f16
__global__ void k_prep_ternary(const float* tern, _Float16* Tab, _Float16* Tba) {
    int idx = blockIdx.x * blockDim.x + threadIdx.x;   // < 2*128*128*8
    int c = idx & 7;
    int b = (idx >> 3) & 127;
    int a = (idx >> 10) & 127;
    int k = idx >> 17;
    _Float16 v = (_Float16)tern[idx];
    Tab[(((k * NH + c) * DM + a) * DM) + b] = v;
    Tba[(((k * NH + c) * DM + b) * DM) + a] = v;
}

// row softmax over D=128: q_z(*mask) -> nz (f32), nz_h (f16), nzT_h (f16)
__global__ void k_softmax_z(const float* src, const int* mask, float* nz,
                            _Float16* nz_h, _Float16* nzT_h) {
    const int row = blockIdx.x;            // n*SEQ + i
    const int n = row >> 8;
    const int i = row & (SEQ - 1);
    const int lane = threadIdx.x;
    const float ms = (mask[row] != 0) ? 1.0f : 0.0f;
    float vals[4];
    float vmax = -3.4e38f;
#pragma unroll
    for (int t = 0; t < 4; ++t) {
        vals[t] = src[row * DM + lane + 32 * t] * ms;
        vmax = fmaxf(vmax, vals[t]);
    }
#pragma unroll
    for (int off = 16; off; off >>= 1) vmax = fmaxf(vmax, __shfl_xor(vmax, off, 32));
    float s = 0.0f;
#pragma unroll
    for (int t = 0; t < 4; ++t) { vals[t] = __expf(vals[t] - vmax); s += vals[t]; }
#pragma unroll
    for (int off = 16; off; off >>= 1) s += __shfl_xor(s, off, 32);
    const float inv = 1.0f / s;
#pragma unroll
    for (int t = 0; t < 4; ++t) {
        const int a = lane + 32 * t;
        const float p = vals[t] * inv;
        nz[row * DM + a] = p;
        nz_h[row * DM + a] = (_Float16)p;
        nzT_h[(n * DM + a) * SEQ + i] = (_Float16)p;
    }
}

// softmax over 128 root logits per batch
__global__ void k_root_softmax(const float* q_root, float* nroot) {
    const int n = blockIdx.x;
    const int lane = threadIdx.x;
    float vals[4];
    float vmax = -3.4e38f;
#pragma unroll
    for (int t = 0; t < 4; ++t) {
        vals[t] = q_root[n * DM + lane + 32 * t];
        vmax = fmaxf(vmax, vals[t]);
    }
#pragma unroll
    for (int off = 16; off; off >>= 1) vmax = fmaxf(vmax, __shfl_xor(vmax, off, 32));
    float s = 0.0f;
#pragma unroll
    for (int t = 0; t < 4; ++t) { vals[t] = __expf(vals[t] - vmax); s += vals[t]; }
#pragma unroll
    for (int off = 16; off; off >>= 1) s += __shfl_xor(s, off, 32);
    const float inv = 1.0f / s;
#pragma unroll
    for (int t = 0; t < 4; ++t) nroot[n * DM + lane + 32 * t] = vals[t] * inv;
}

// v[n][a][c] = sum_r rt[a,r,c] * nroot[n,r]
__global__ void k_root_v(const float* rt, const float* nroot, float* v) {
    const int n = blockIdx.x;
    const int a = threadIdx.x;   // 128 threads
    float acc[NH] = {};
    for (int r = 0; r < DM; ++r) {
        const float nr = nroot[n * DM + r];
        const float* rp = rt + (a * DM + r) * NH;
#pragma unroll
        for (int c = 0; c < NH; ++c) acc[c] += rp[c] * nr;
    }
#pragma unroll
    for (int c = 0; c < NH; ++c) v[(n * DM + a) * NH + c] = acc[c];
}

// s_root[n][c][i] = sum_a nz[n,i,a] * v[n,a,c]
__global__ void k_s_root(const float* nz, const float* v, float* s_root) {
    const int idx = blockIdx.x * blockDim.x + threadIdx.x;   // < B*NH*SEQ
    const int i = idx & (SEQ - 1);
    const int c = (idx >> 8) & 7;
    const int n = idx >> 11;
    float acc = 0.0f;
    for (int a = 0; a < DM; ++a)
        acc += nz[(n * SEQ + i) * DM + a] * v[(n * DM + a) * NH + c];
    s_root[(n * NH + c) * SEQ + i] = acc;
}

// ---------------- WMMA GEMM kernels -----------------------------------------

// g[k][n][c] (SEQ x DM) = nz[n] @ T_k[:, :, c]   (K = a = DM)
// block: 8 waves share A strip nz[n][tm..tm+16)  (staged in LDS)
__global__ void k_gemm_g(const _Float16* nz_h, const _Float16* Tba, _Float16* g) {
    __shared__ __align__(16) _Float16 sA[16 * DM];           // 4 KB
    const int blk = blockIdx.x;           // mat*16 + tm16
    const int mat = blk >> 4;             // k*64 + n*8 + c
    const int tm = (blk & 15) << 4;
    const int c = mat & 7;
    const int n = (mat >> 3) & 7;
    const int k = mat >> 6;
    const _Float16* Astrip = nz_h + (size_t)n * SEQ * DM + (size_t)tm * DM;
    cp16(Astrip + threadIdx.x * 8, sA + threadIdx.x * 8);    // 256 lanes * 16B
    stage_join();
    const int tn = (threadIdx.x >> 5) << 4;
    const _Float16* Bp = Tba + (size_t)((k * NH + c) * DM + tn) * DM;
    v8f acc = {};
#pragma unroll
    for (int kk = 0; kk < DM; kk += 32)
        acc = wmma_f16(load_frag(sA + kk, DM), load_frag(Bp + kk, DM), acc);
    _Float16* out = g + (size_t)((k * BATCH + n) * NH + c) * SEQ * DM
                      + (size_t)tm * DM + tn;
    store_tile_f16(out, DM, acc);
}

// s_word[n][c][i][j] = (j>i)?P0:(j<i)?P1:0 with P_k = g_k @ nz^T (K = b = DM)
// block: 8 waves share g0/g1 strips, cover 128 of 256 j columns
__global__ void k_gemm_P(const _Float16* g, const _Float16* nz_h, float* s_word) {
    __shared__ __align__(16) _Float16 sA[2][16 * DM];        // 8 KB
    const int blk = blockIdx.x;           // (nc*16 + tm16)*2 + jh
    const int jh = blk & 1;
    const int tm = ((blk >> 1) & 15) << 4;
    const int nc = blk >> 5;              // n*8 + c
    const int n = nc >> 3;
    const _Float16* A0 = g + (size_t)nc * SEQ * DM + (size_t)tm * DM;
    const _Float16* A1 = A0 + (size_t)BATCH * NH * SEQ * DM;
    cp16(A0 + threadIdx.x * 8, sA[0] + threadIdx.x * 8);
    cp16(A1 + threadIdx.x * 8, sA[1] + threadIdx.x * 8);
    stage_join();
    const int tn = (jh << 7) + ((threadIdx.x >> 5) << 4);
    const _Float16* Bp = nz_h + (size_t)n * SEQ * DM + (size_t)tn * DM; // rows j
    v8f acc0 = {}, acc1 = {};
#pragma unroll
    for (int kk = 0; kk < DM; kk += 32) {
        v16h b = load_frag(Bp + kk, DM);
        acc0 = wmma_f16(load_frag(sA[0] + kk, DM), b, acc0);
        acc1 = wmma_f16(load_frag(sA[1] + kk, DM), b, acc1);
    }
    float* out = s_word + (size_t)(nc * SEQ + tm) * SEQ + tn;
    const int lane = threadIdx.x & 31;
    const int nn = lane & 15;
    const int h = lane >> 4;
#pragma unroll
    for (int r = 0; r < 8; ++r) {
        const int i = tm + 8 * h + r;
        const int j = tn + nn;
        float val = (j > i) ? acc0[r] : ((j < i) ? acc1[r] : 0.0f);
        out[(8 * h + r) * SEQ + nn] = val;
    }
}

// softmax over 257 cols (s_root | s_word row) with bad_h -> NEG.
// Emits nh_r plus triangular-masked f16 copies in both orientations.
__global__ void k_head_softmax(const float* s_word, const float* s_root, const int* mask,
                               float* nh_r, _Float16* nhU, _Float16* nhL,
                               _Float16* nhUt, _Float16* nhLt) {
    const int row = blockIdx.x;                 // (n*NH + c)*SEQ + i
    const int i = row & (SEQ - 1);
    const int nc = row >> 8;
    const int n = nc >> 3;
    const int lane = threadIdx.x;
    const float* sw = s_word + (size_t)row * SEQ;
    const bool mi = mask[n * SEQ + i] != 0;
    float vals[8];
    float vmax = NEGV;
#pragma unroll
    for (int t = 0; t < 8; ++t) {
        const int j = lane + 32 * t;
        const bool mj = mask[n * SEQ + j] != 0;
        float vv = (mi && mj && (j != i)) ? sw[j] : NEGV;
        vals[t] = vv;
        vmax = fmaxf(vmax, vv);
    }
    const float v0 = mi ? s_root[row] : NEGV;
    vmax = fmaxf(vmax, v0);
#pragma unroll
    for (int off = 16; off; off >>= 1) vmax = fmaxf(vmax, __shfl_xor(vmax, off, 32));
    float s = 0.0f;
#pragma unroll
    for (int t = 0; t < 8; ++t) { vals[t] = __expf(vals[t] - vmax); s += vals[t]; }
#pragma unroll
    for (int off = 16; off; off >>= 1) s += __shfl_xor(s, off, 32);
    const float e0 = __expf(v0 - vmax);
    const float inv = 1.0f / (s + e0);
    if (lane == 0) nh_r[row] = e0 * inv;
    const size_t base = (size_t)nc * SEQ * SEQ;
#pragma unroll
    for (int t = 0; t < 8; ++t) {
        const int j = lane + 32 * t;
        const float p = vals[t] * inv;
        const _Float16 pu = (j > i) ? (_Float16)p : (_Float16)0.0f;
        const _Float16 pl = (j < i) ? (_Float16)p : (_Float16)0.0f;
        nhU[base + (size_t)i * SEQ + j] = pu;
        nhL[base + (size_t)i * SEQ + j] = pl;
        nhUt[base + (size_t)j * SEQ + i] = pu;
        nhLt[base + (size_t)j * SEQ + i] = pl;
    }
}

// Y0/Y1/Z0/Z1 (SEQ x DM) = nh_variant (SEQ x SEQ) @ nz (via nzT as NxK), K=SEQ
// block: 8 waves share the nh strip (staged), each owns one a-tile
__global__ void k_gemm_yz(const _Float16* nhU, const _Float16* nhL,
                          const _Float16* nhUt, const _Float16* nhLt,
                          const _Float16* nzT_h,
                          _Float16* Y0, _Float16* Y1, _Float16* Z0, _Float16* Z1) {
    __shared__ __align__(16) _Float16 sA[16 * SEQ];          // 8 KB
    const int blk = blockIdx.x;           // (var*64 + nc)*16 + tm16
    const int tm = (blk & 15) << 4;
    const int nc = (blk >> 4) & 63;
    const int var = blk >> 10;            // 0..3
    const int n = nc >> 3;
    const _Float16* srcs[4] = { nhU, nhL, nhUt, nhLt };
    _Float16* dsts[4] = { Y0, Y1, Z0, Z1 };
    const _Float16* A = srcs[var] + (size_t)nc * SEQ * SEQ + (size_t)tm * SEQ;
    cp16(A + threadIdx.x * 8, sA + threadIdx.x * 8);
    cp16(A + 2048 + threadIdx.x * 8, sA + 2048 + threadIdx.x * 8);
    stage_join();
    const int tn = (threadIdx.x >> 5) << 4;
    const _Float16* Bp = nzT_h + (size_t)n * DM * SEQ + (size_t)tn * SEQ;
    v8f acc = {};
#pragma unroll
    for (int kk = 0; kk < SEQ; kk += 32)
        acc = wmma_f16(load_frag(sA + kk, SEQ), load_frag(Bp + kk, SEQ), acc);
    _Float16* out = dsts[var] + (size_t)nc * SEQ * DM + (size_t)tm * DM + tn;
    store_tile_f16(out, DM, acc);
}

// m_dep[n] = sum_c ( Y0 @ Tab[0][c] (as NxK) + Y1 @ Tab[1][c] ), K=b=DM
// block: 8 waves share all 16 Y strips (64 KB staged), each owns one a-tile
__global__ void k_gemm_mdep(const _Float16* Y0, const _Float16* Y1,
                            const _Float16* Tab, float* m_dep) {
    __shared__ __align__(16) _Float16 sY[16][16 * DM];       // 64 KB
    const int blk = blockIdx.x;           // n*16 + tm16
    const int n = blk >> 4;
    const int tm = (blk & 15) << 4;
#pragma unroll
    for (int q = 0; q < 16; ++q) {        // q = c*2 + k2
        const _Float16* A = ((q & 1) ? Y1 : Y0)
            + (size_t)(n * NH + (q >> 1)) * SEQ * DM + (size_t)tm * DM;
        cp16(A + threadIdx.x * 8, sY[q] + threadIdx.x * 8);
    }
    stage_join();
    const int tn = (threadIdx.x >> 5) << 4;
    v8f acc = {};
    for (int q = 0; q < 16; ++q) {
        const int c = q >> 1, k2 = q & 1;
        const _Float16* Bp = Tab + (size_t)((k2 * NH + c) * DM + tn) * DM;
#pragma unroll
        for (int kk = 0; kk < DM; kk += 32)
            acc = wmma_f16(load_frag(sY[q] + kk, DM), load_frag(Bp + kk, DM), acc);
    }
    store_tile_f32(m_dep + (size_t)n * SEQ * DM + (size_t)tm * DM + tn, DM, acc);
}

// m_head[n] = sum_c ( Z0 @ Tba[0][c] (as NxK) + Z1 @ Tba[1][c] ), K=a=DM
__global__ void k_gemm_mhead(const _Float16* Z0, const _Float16* Z1,
                             const _Float16* Tba, float* m_head) {
    __shared__ __align__(16) _Float16 sZ[16][16 * DM];       // 64 KB
    const int blk = blockIdx.x;
    const int n = blk >> 4;
    const int tm = (blk & 15) << 4;
#pragma unroll
    for (int q = 0; q < 16; ++q) {
        const _Float16* A = ((q & 1) ? Z1 : Z0)
            + (size_t)(n * NH + (q >> 1)) * SEQ * DM + (size_t)tm * DM;
        cp16(A + threadIdx.x * 8, sZ[q] + threadIdx.x * 8);
    }
    stage_join();
    const int tn = (threadIdx.x >> 5) << 4;
    v8f acc = {};
    for (int q = 0; q < 16; ++q) {
        const int c = q >> 1, k2 = q & 1;
        const _Float16* Bp = Tba + (size_t)((k2 * NH + c) * DM + tn) * DM;
#pragma unroll
        for (int kk = 0; kk < DM; kk += 32)
            acc = wmma_f16(load_frag(sZ[q] + kk, DM), load_frag(Bp + kk, DM), acc);
    }
    store_tile_f32(m_head + (size_t)n * SEQ * DM + (size_t)tm * DM + tn, DM, acc);
}

// q_z_out = (x + m_dep + m_head + m_root) * mask, m_root = sum_c nh_r * v
__global__ void k_sz(const float* x, const float* m_dep, const float* m_head,
                     const float* nh_r, const float* v, const int* mask, float* out) {
    const int idx = blockIdx.x * blockDim.x + threadIdx.x;  // < B*SEQ*DM
    const int a = idx & (DM - 1);
    const int i = (idx >> 7) & (SEQ - 1);
    const int n = idx >> 15;
    float mr = 0.0f;
#pragma unroll
    for (int c = 0; c < NH; ++c)
        mr += nh_r[(n * NH + c) * SEQ + i] * v[(n * DM + a) * NH + c];
    float val = x[idx] + m_dep[idx] + m_head[idx] + mr;
    val *= (mask[n * SEQ + i] != 0) ? 1.0f : 0.0f;
    out[idx] = val;
}

// t[n][a][c] = sum_i nh_r[n,c,i] * nz[n,i,a]
__global__ void k_t(const float* nh_r, const float* nz, float* t) {
    const int n = blockIdx.x;
    const int a = threadIdx.x;   // 128
    float acc[NH] = {};
    for (int i = 0; i < SEQ; ++i) {
        const float nzv = nz[(n * SEQ + i) * DM + a];
#pragma unroll
        for (int c = 0; c < NH; ++c) acc[c] += nh_r[(n * NH + c) * SEQ + i] * nzv;
    }
#pragma unroll
    for (int c = 0; c < NH; ++c) t[(n * DM + a) * NH + c] = acc[c];
}

// q_root[n][r] = sum_{a,c} t[n,a,c] * rt[a,r,c]
__global__ void k_sr(const float* t, const float* rt, float* q_root) {
    const int n = blockIdx.x;
    const int r = threadIdx.x;   // 128
    float acc = 0.0f;
    for (int a = 0; a < DM; ++a) {
        const float* tp = t + (n * DM + a) * NH;
        const float* rp = rt + (a * DM + r) * NH;
#pragma unroll
        for (int c = 0; c < NH; ++c) acc += tp[c] * rp[c];
    }
    q_root[n * DM + r] = acc;
}

// ---------------------------------------------------------------------------

extern "C" void kernel_launch(void* const* d_in, const int* in_sizes, int n_in,
                              void* d_out, int out_size, void* d_ws, size_t ws_size,
                              hipStream_t stream) {
    (void)in_sizes; (void)n_in; (void)out_size; (void)ws_size;
    const float* x    = (const float*)d_in[0];
    const int*   mask = (const int*)d_in[1];
    const float* tern = (const float*)d_in[2];
    const float* rt   = (const float*)d_in[3];
    float* outp = (float*)d_out;

    char* p = (char*)d_ws;
    auto alloc = [&](size_t bytes) -> char* {
        char* r = p;
        p += (bytes + 255) & ~(size_t)255;
        return r;
    };

    float*    q_z    = (float*)alloc((size_t)BATCH * SEQ * DM * 4);
    float*    q_root = (float*)alloc((size_t)BATCH * DM * 4);
    float*    nz     = (float*)alloc((size_t)BATCH * SEQ * DM * 4);
    float*    nroot  = (float*)alloc((size_t)BATCH * DM * 4);
    float*    vbuf   = (float*)alloc((size_t)BATCH * DM * NH * 4);
    float*    s_root = (float*)alloc((size_t)BATCH * NH * SEQ * 4);
    float*    nh_r   = (float*)alloc((size_t)BATCH * NH * SEQ * 4);
    float*    tbuf   = (float*)alloc((size_t)BATCH * DM * NH * 4);
    float*    m_dep  = (float*)alloc((size_t)BATCH * SEQ * DM * 4);
    float*    m_head = (float*)alloc((size_t)BATCH * SEQ * DM * 4);
    float*    s_word = (float*)alloc((size_t)BATCH * NH * SEQ * SEQ * 4);
    _Float16* nz_h   = (_Float16*)alloc((size_t)BATCH * SEQ * DM * 2);
    _Float16* nzT_h  = (_Float16*)alloc((size_t)BATCH * DM * SEQ * 2);
    _Float16* Tab    = (_Float16*)alloc((size_t)NKD * NH * DM * DM * 2);
    _Float16* Tba    = (_Float16*)alloc((size_t)NKD * NH * DM * DM * 2);
    _Float16* gbuf   = (_Float16*)alloc((size_t)NKD * BATCH * NH * SEQ * DM * 2);
    _Float16* nhU    = (_Float16*)alloc((size_t)BATCH * NH * SEQ * SEQ * 2);
    _Float16* nhL    = (_Float16*)alloc((size_t)BATCH * NH * SEQ * SEQ * 2);
    _Float16* nhUt   = (_Float16*)alloc((size_t)BATCH * NH * SEQ * SEQ * 2);
    _Float16* nhLt   = (_Float16*)alloc((size_t)BATCH * NH * SEQ * SEQ * 2);
    _Float16* Y0     = (_Float16*)alloc((size_t)BATCH * NH * SEQ * DM * 2);
    _Float16* Y1     = (_Float16*)alloc((size_t)BATCH * NH * SEQ * DM * 2);
    _Float16* Z0     = (_Float16*)alloc((size_t)BATCH * NH * SEQ * DM * 2);
    _Float16* Z1     = (_Float16*)alloc((size_t)BATCH * NH * SEQ * DM * 2);

    // one-time prep
    hipLaunchKernelGGL(k_prep_ternary, dim3(1024), dim3(256), 0, stream, tern, Tab, Tba);
    hipLaunchKernelGGL(k_zero, dim3(4), dim3(256), 0, stream, q_root, BATCH * DM);

    for (int iter = 0; iter < 2; ++iter) {
        const float* zsrc = (iter == 0) ? x : q_z;
        float* zdst = (iter == 0) ? q_z : outp;

        hipLaunchKernelGGL(k_softmax_z, dim3(BATCH * SEQ), dim3(32), 0, stream,
                           zsrc, mask, nz, nz_h, nzT_h);
        hipLaunchKernelGGL(k_root_softmax, dim3(BATCH), dim3(32), 0, stream, q_root, nroot);
        hipLaunchKernelGGL(k_root_v, dim3(BATCH), dim3(DM), 0, stream, rt, nroot, vbuf);
        hipLaunchKernelGGL(k_s_root, dim3(64), dim3(256), 0, stream, nz, vbuf, s_root);

        // G = nz @ T : 128 matrices x 16 strips = 2048 blocks (8 waves each)
        hipLaunchKernelGGL(k_gemm_g, dim3(2048), dim3(256), 0, stream, nz_h, Tba, gbuf);
        // s_word via P0/P1 : 64 nc x 16 strips x 2 j-halves = 2048 blocks
        hipLaunchKernelGGL(k_gemm_P, dim3(2048), dim3(256), 0, stream, gbuf, nz_h, s_word);
        hipLaunchKernelGGL(k_head_softmax, dim3(BATCH * NH * SEQ), dim3(32), 0, stream,
                           s_word, s_root, mask, nh_r, nhU, nhL, nhUt, nhLt);
        // Y0,Y1,Z0,Z1 : 4 variants x 64 nc x 16 strips = 4096 blocks
        hipLaunchKernelGGL(k_gemm_yz, dim3(4096), dim3(256), 0, stream,
                           nhU, nhL, nhUt, nhLt, nzT_h, Y0, Y1, Z0, Z1);
        // m_dep / m_head : 8 n x 16 strips = 128 blocks
        hipLaunchKernelGGL(k_gemm_mdep, dim3(128), dim3(256), 0, stream, Y0, Y1, Tab, m_dep);
        hipLaunchKernelGGL(k_gemm_mhead, dim3(128), dim3(256), 0, stream, Z0, Z1, Tba, m_head);
        hipLaunchKernelGGL(k_sz, dim3(1024), dim3(256), 0, stream,
                           x, m_dep, m_head, nh_r, vbuf, mask, zdst);

        if (iter == 0) {
            hipLaunchKernelGGL(k_t, dim3(BATCH), dim3(DM), 0, stream, nh_r, nz, tbuf);
            hipLaunchKernelGGL(k_sr, dim3(BATCH), dim3(DM), 0, stream, tbuf, rt, q_root);
        }
    }
}
